// _SelfAttention2d_59227599012029
// MI455X (gfx1250) — compile-verified, matrix operand order checked by reading, not run on hardware
//
#include <hip/hip_runtime.h>

typedef __attribute__((ext_vector_type(16))) __bf16 v16bf;
typedef __attribute__((ext_vector_type(8)))  float  v8f;
typedef __attribute__((ext_vector_type(4)))  unsigned int u32x4;

#define C_ 256
#define N_ 4096
#define B_ 8

union Frag { u32x4 q[2]; v16bf v; };

__device__ __forceinline__ unsigned short f2bf(float f) {
  unsigned int u = __builtin_bit_cast(unsigned int, f);
  unsigned int r = u + 0x7FFFu + ((u >> 16) & 1u);  // round-to-nearest-even
  return (unsigned short)(r >> 16);
}

// Load one 16x32 bf16 A/B fragment slice for this lane as two b128 loads.
__device__ __forceinline__ Frag ldfrag(const __bf16* __restrict__ p, int elemOff) {
  Frag f;
  f.q[0] = *(const u32x4*)(p + elemOff);
  f.q[1] = *(const u32x4*)(p + elemOff + 16);
  return f;
}

__device__ __forceinline__ Frag ldfrag_lds(const __bf16* p) {
  Frag f;
  f.q[0] = *(const u32x4*)p;
  f.q[1] = *(const u32x4*)(p + 16);
  return f;
}

__device__ __forceinline__ v8f wmma_bf16(Frag a, Frag b, v8f c) {
  return __builtin_amdgcn_wmma_f32_16x16x32_bf16(false, a.v, false, b.v, (short)0, c, false, false);
}

// Async global->LDS copy of 16 bytes (ASYNCcnt-tracked, CDNA5).
// LDS address = low 32 bits of the generic pointer (workgroup-relative offset).
__device__ __forceinline__ void async_cp16(const __bf16* g, __bf16* l) {
  unsigned lo = (unsigned)(size_t)l;
  asm volatile("global_load_async_to_lds_b128 %0, %1, off" :: "v"(lo), "v"(g) : "memory");
}
__device__ __forceinline__ void wait_async0() {
  asm volatile("s_wait_asynccnt 0x0" ::: "memory");
}

// ---------------- GroupNorm stats: one block per (b, group) ----------------
__global__ void gn_stats_kernel(const float* __restrict__ x, float* __restrict__ stats) {
  int bg = blockIdx.x;                 // 0..63
  int b = bg >> 3, g = bg & 7;
  const float* px = x + ((size_t)b * C_ + g * 32) * N_;
  float s = 0.f, ss = 0.f;
  for (int i = threadIdx.x; i < 32 * N_; i += 256) {
    float v = px[i];
    s += v; ss += v * v;
  }
  __shared__ float r1[256], r2[256];
  r1[threadIdx.x] = s; r2[threadIdx.x] = ss;
  __syncthreads();
  for (int off = 128; off > 0; off >>= 1) {
    if (threadIdx.x < (unsigned)off) {
      r1[threadIdx.x] += r1[threadIdx.x + off];
      r2[threadIdx.x] += r2[threadIdx.x + off];
    }
    __syncthreads();
  }
  if (threadIdx.x == 0) {
    float inv = 1.0f / (32.f * (float)N_);
    float mu = r1[0] * inv;
    float var = r2[0] * inv - mu * mu;
    stats[bg * 2]     = mu;
    stats[bg * 2 + 1] = rsqrtf(var + 1e-5f);
  }
}

// ------- GroupNorm apply + transpose: x[b,c,n] f32 -> yT[b,n,c] bf16 -------
__global__ void gn_apply_kernel(const float* __restrict__ x, const float* __restrict__ gw,
                                const float* __restrict__ gb, const float* __restrict__ stats,
                                __bf16* __restrict__ yT) {
  int blk = blockIdx.x;
  int nt = blk & 127;
  int ct = (blk >> 7) & 7;
  int b  = blk >> 10;
  int c0 = ct * 32, n0 = nt * 32;
  __shared__ float tile[32][33];
  int lx = threadIdx.x & 31, ly = threadIdx.x >> 5;
  const float* px = x + (size_t)b * C_ * N_;
#pragma unroll
  for (int r = 0; r < 4; ++r) {
    int c = ly + r * 8;
    tile[c][lx] = px[(size_t)(c0 + c) * N_ + n0 + lx];
  }
  __syncthreads();
  int c = c0 + lx;
  int g = c >> 5;
  float mu = stats[(b * 8 + g) * 2];
  float rs = stats[(b * 8 + g) * 2 + 1];
  float w  = gw[c] * rs;
  float bb = gb[c] - mu * w;
  __bf16* py = yT + (size_t)b * N_ * C_;
#pragma unroll
  for (int r = 0; r < 4; ++r) {
    int n = ly + r * 8;
    py[(size_t)(n0 + n) * C_ + c] = __builtin_bit_cast(__bf16, f2bf(tile[lx][n] * w + bb));
  }
}

// ---------------- f32 -> bf16 weight conversion ----------------
__global__ void wconv_kernel(const float* __restrict__ w, __bf16* __restrict__ o, int n) {
  int i = blockIdx.x * 256 + threadIdx.x;
  if (i < n) o[i] = __builtin_bit_cast(__bf16, f2bf(w[i]));
}

// ------------- Projection GEMM: acc[m,o] = sum_k A[m,k] * W[o,k] -------------
__global__ void proj_kernel(const __bf16* __restrict__ A, const __bf16* __restrict__ W,
                            __bf16* __restrict__ outB, float* __restrict__ outF,
                            const float* __restrict__ x, const float* __restrict__ bp,
                            int mode) {
  int wave = (blockIdx.x * blockDim.x + threadIdx.x) >> 5;
  int lane = threadIdx.x & 31;
  int half = lane >> 4, l15 = lane & 15;
  int nt = wave & 15;
  int mt = (wave >> 4) & 255;
  int b  = wave >> 12;
  int m0 = mt * 16, o0 = nt * 16;
  const __bf16* Ap = A + (size_t)b * N_ * C_;
  v8f acc = {};
#pragma unroll 1
  for (int kc = 0; kc < 8; ++kc) {
    Frag af = ldfrag(Ap, (m0 + l15) * C_ + kc * 32 + half * 8);
    Frag bf = ldfrag(W,  (o0 + l15) * C_ + kc * 32 + half * 8);
    acc = wmma_bf16(af, bf, acc);
  }
  if (mode == 0) {
    __bf16* p = outB + (size_t)b * N_ * C_;
#pragma unroll
    for (int j = 0; j < 8; ++j) {
      int row = j + 8 * half;
      p[(size_t)(m0 + row) * C_ + o0 + l15] = __builtin_bit_cast(__bf16, f2bf(acc[j]));
    }
  } else if (mode == 1) {
    __bf16* p = outB + (size_t)b * C_ * N_;
#pragma unroll
    for (int j = 0; j < 8; ++j) {
      int row = j + 8 * half;
      p[(size_t)(o0 + l15) * N_ + m0 + row] = __builtin_bit_cast(__bf16, f2bf(acc[j]));
    }
  } else {
    float* p = outF + (size_t)b * C_ * N_;
    const float* pxx = x + (size_t)b * C_ * N_;
    float bias = bp[o0 + l15];
#pragma unroll
    for (int j = 0; j < 8; ++j) {
      int row = j + 8 * half;
      size_t idx = (size_t)(o0 + l15) * N_ + m0 + row;
      p[idx] = pxx[idx] + bias + acc[j];
    }
  }
}

// -------- Flash attention: block = 4 waves x 16 query rows; K/V tiles staged --------
// -------- to LDS with double-buffered async copies shared by all 4 waves   --------
__global__ void flash_kernel(const __bf16* __restrict__ Q, const __bf16* __restrict__ K,
                             const __bf16* __restrict__ Vt, __bf16* __restrict__ O) {
  __shared__ __align__(16) __bf16 kbuf[2][32 * 256];        // 2 x 16 KB
  __shared__ __align__(16) __bf16 vbuf[2][256 * 32];        // 2 x 16 KB
  __shared__ __align__(16) unsigned short pshare[4][16 * 32];  // 4 KB
  int tid  = threadIdx.x;
  int widx = tid >> 5, lane = tid & 31;
  int half = lane >> 4, l15 = lane & 15;
  int blk = blockIdx.x;            // b*64 + qgroup
  int qg = blk & 63, b = blk >> 6;
  int q0 = qg * 64 + widx * 16;
  const __bf16* Qp = Q + (size_t)b * N_ * C_;
  const __bf16* Kp = K + (size_t)b * N_ * C_;
  const __bf16* Vp = Vt + (size_t)b * C_ * N_;
  unsigned short* ps = pshare[widx];

  // Stage one 32-wide KV tile into LDS buffer `buf` (whole block cooperates).
  auto stage = [&](int buf, int kv) {
#pragma unroll
    for (int r = 0; r < 8; ++r) {
      int e = (tid + r * 128) * 8;          // K tile: 16 KB fully contiguous
      async_cp16(Kp + (size_t)kv * C_ + e, &kbuf[buf][e]);
      int qd = tid + r * 128;               // V tile: 256 rows x 64 B
      int c = qd >> 2, part = qd & 3;
      async_cp16(Vp + (size_t)c * N_ + kv + part * 8, &vbuf[buf][c * 32 + part * 8]);
    }
  };

  // Q tile A-fragments resident in VGPRs
  Frag qf[8];
#pragma unroll
  for (int kc = 0; kc < 8; ++kc)
    qf[kc] = ldfrag(Qp, (q0 + l15) * C_ + kc * 32 + half * 8);

  v8f oacc[16] = {};
  float mst[8], lst[8];
#pragma unroll
  for (int j = 0; j < 8; ++j) { mst[j] = -1e30f; lst[j] = 0.f; }

  const float scale = 0.0625f;  // C^-0.5

  stage(0, 0);
  wait_async0();
  __syncthreads();

  for (int it = 0; it < N_ / 32; ++it) {
    int kv = it * 32;
    int cur = it & 1;
    if (it + 1 < N_ / 32) stage(1 - cur, kv + 32);  // prefetch next tile

    // S = Q * K^T, K fragments from LDS (ds_load_b128)
    v8f s0 = {}, s1 = {};
#pragma unroll
    for (int kc = 0; kc < 8; ++kc) {
      Frag b0 = ldfrag_lds(&kbuf[cur][l15 * 256 + kc * 32 + half * 8]);
      Frag b1 = ldfrag_lds(&kbuf[cur][(16 + l15) * 256 + kc * 32 + half * 8]);
      s0 = wmma_bf16(qf[kc], b0, s0);
      s1 = wmma_bf16(qf[kc], b1, s1);
    }

    // Online softmax; row j lives on a 16-lane half (C/D layout)
#pragma unroll
    for (int j = 0; j < 8; ++j) {
      float v0 = s0[j] * scale, v1 = s1[j] * scale;
      float mx = fmaxf(v0, v1);
#pragma unroll
      for (int m = 1; m < 16; m <<= 1) mx = fmaxf(mx, __shfl_xor(mx, m, 32));
      float mn = fmaxf(mst[j], mx);
      float alpha = __expf(mst[j] - mn);
      mst[j] = mn;
      float p0 = __expf(v0 - mn), p1 = __expf(v1 - mn);
      float rs = p0 + p1;
#pragma unroll
      for (int m = 1; m < 16; m <<= 1) rs += __shfl_xor(rs, m, 32);
      lst[j] = lst[j] * alpha + rs;
#pragma unroll
      for (int g = 0; g < 16; ++g) oacc[g][j] *= alpha;
      int row = j + 8 * half;
      ps[row * 32 + l15]      = f2bf(p0);
      ps[row * 32 + 16 + l15] = f2bf(p1);
    }
    asm volatile("" ::: "memory");  // DS in-order within wave; block TBAA reorder

    // P A-fragment from LDS (D-layout -> A-layout transpose)
    Frag pf;
    pf.q[0] = *(const u32x4*)(ps + l15 * 32 + half * 8);
    pf.q[1] = *(const u32x4*)(ps + l15 * 32 + half * 8 + 16);
    asm volatile("" ::: "memory");

    // O += P(16x32) * V(32x16), V fragments from LDS
#pragma unroll
    for (int g = 0; g < 16; ++g) {
      Frag vf = ldfrag_lds(&vbuf[cur][(g * 16 + l15) * 32 + half * 8]);
      oacc[g] = wmma_bf16(pf, vf, oacc[g]);
    }

    wait_async0();     // own prefetch done (overlapped with compute above)
    __syncthreads();   // whole block's prefetch done; tile consumed by all waves
  }

  // Normalize by l and write O bf16 [B,N,C]
  __bf16* Op = O + (size_t)b * N_ * C_;
#pragma unroll
  for (int j = 0; j < 8; ++j) {
    int row = j + 8 * half;
    float inv = 1.0f / lst[j];
#pragma unroll
    for (int g = 0; g < 16; ++g)
      Op[(size_t)(q0 + row) * C_ + g * 16 + l15] =
          __builtin_bit_cast(__bf16, f2bf(oacc[g][j] * inv));
  }
}

extern "C" void kernel_launch(void* const* d_in, const int* in_sizes, int n_in,
                              void* d_out, int out_size, void* d_ws, size_t ws_size,
                              hipStream_t stream) {
  (void)in_sizes; (void)n_in; (void)out_size; (void)ws_size;
  const float* x    = (const float*)d_in[0];
  const float* gn_w = (const float*)d_in[1];
  const float* gn_b = (const float*)d_in[2];
  const float* wq   = (const float*)d_in[3];
  const float* wk   = (const float*)d_in[4];
  const float* wv   = (const float*)d_in[5];
  const float* wp   = (const float*)d_in[6];
  const float* bp   = (const float*)d_in[7];
  float* out = (float*)d_out;
  char* ws = (char*)d_ws;

  const size_t szTok = (size_t)B_ * N_ * C_ * sizeof(__bf16);  // 16 MiB
  float*  stats = (float*)ws;
  __bf16* wqb = (__bf16*)(ws + 1024);
  __bf16* wkb = wqb + C_ * C_;
  __bf16* wvb = wkb + C_ * C_;
  __bf16* wpb = wvb + C_ * C_;
  __bf16* yT  = (__bf16*)(ws + 1024 + 4 * C_ * C_ * sizeof(__bf16));
  __bf16* Qb  = (__bf16*)((char*)yT + szTok);
  __bf16* Kb  = (__bf16*)((char*)Qb + szTok);
  __bf16* Vb  = (__bf16*)((char*)Kb + szTok);
  __bf16* Ob  = yT;  // yT dead after QKV projections; reuse for O

  wconv_kernel<<<256, 256, 0, stream>>>(wq, wqb, C_ * C_);
  wconv_kernel<<<256, 256, 0, stream>>>(wk, wkb, C_ * C_);
  wconv_kernel<<<256, 256, 0, stream>>>(wv, wvb, C_ * C_);
  wconv_kernel<<<256, 256, 0, stream>>>(wp, wpb, C_ * C_);

  gn_stats_kernel<<<64, 256, 0, stream>>>(x, stats);
  gn_apply_kernel<<<8192, 256, 0, stream>>>(x, gn_w, gn_b, stats, yT);

  proj_kernel<<<8192, 128, 0, stream>>>(yT, wqb, Qb, nullptr, nullptr, nullptr, 0);
  proj_kernel<<<8192, 128, 0, stream>>>(yT, wkb, Kb, nullptr, nullptr, nullptr, 0);
  proj_kernel<<<8192, 128, 0, stream>>>(yT, wvb, Vb, nullptr, nullptr, nullptr, 1);

  flash_kernel<<<512, 128, 0, stream>>>(Qb, Kb, Vb, Ob);

  proj_kernel<<<8192, 128, 0, stream>>>(Ob, wpb, nullptr, out, x, bp, 2);
}
